// MoE_71098888618613
// MI455X (gfx1250) — compile-verified, matrix operand order checked by reading, not run on hardware
//
#include <hip/hip_runtime.h>
#include <hip/hip_bf16.h>
#include <stdint.h>

typedef __attribute__((ext_vector_type(16))) __bf16 v16bf;
typedef __attribute__((ext_vector_type(8)))  float  v8f;
typedef __attribute__((ext_vector_type(4)))  unsigned int v4u;
typedef __attribute__((ext_vector_type(8)))  int v8i;
typedef __attribute__((ext_vector_type(4)))  int v4i;

#define B_TOK   16384
#define IN_DIM  1024
#define NEXP    8
#define HID     128
#define NCLS    1024

// workspace byte offsets
#define XB_OFF   0UL            // bf16 x            33,554,432 B
#define W1S_OFF  33554432UL     // swizzled w1 bf16   2,097,152 B
#define W2S_OFF  35651584UL     // swizzled w2 bf16   2,097,152 B
#define AT_OFF   37748736UL     // assign_token int     524,288 B
#define AG_OFF   38273024UL     // assign_gate  f32     524,288 B
#define TV_OFF   38797312UL     // top_vals [2][B]      131,072 B
#define CNT_OFF  38928384UL     // expert counts            32 B
#define GSUM_OFF 38928416UL     // 2 floats

#if defined(__has_builtin)
# if __has_builtin(__builtin_amdgcn_tensor_load_to_lds)
#  define HAVE_TDM 1
# endif
#endif
#ifndef HAVE_TDM
# define HAVE_TDM 0
#endif

__device__ inline v8f v8f_zero() {
  v8f z;
#pragma unroll
  for (int i = 0; i < 8; ++i) z[i] = 0.f;
  return z;
}

#if HAVE_TDM
// 1-D TDM copy of 8KB (1024 x 8-byte units) global -> LDS.
// D# group0: count=1 | lds_addr | global_addr | type=2
// D# group1: data_size=3 (8B); tensor_dim0=tile_dim0=1024; stride0=1024
// Groups 2/3 unused (<=2-D tensor); 6-arg builtin (clang-23 / therock-10.0).
__device__ inline void tdm_load_8k(const __bf16* gsrc, unsigned lds_off) {
  unsigned long long ga = (unsigned long long)(uintptr_t)gsrc;
  v4u g0;
  g0[0] = 1u;                                                  // count=1
  g0[1] = lds_off;                                             // LDS byte addr
  g0[2] = (unsigned)ga;                                        // global addr lo
  g0[3] = ((unsigned)(ga >> 32) & 0x01FFFFFFu) | 0x80000000u;  // addr hi | type=2
  v8i g1;
  g1[0] = (int)(3u << 16);        // data_size = 8 bytes
  g1[1] = (int)(1024u << 16);     // tensor_dim0[15:0]  -> bits[63:48]
  g1[2] = 0;                      // tensor_dim0 hi / tensor_dim1 lo
  g1[3] = (int)(1024u << 16);     // tile_dim0 = 1024   -> bits[127:112]
  g1[4] = 0;                      // tile_dim1 = tile_dim2 = 0 (1-D)
  g1[5] = 1024;                   // tensor_dim0_stride
  g1[6] = 0; g1[7] = 0;
  v4i z4; z4[0] = z4[1] = z4[2] = z4[3] = 0;
  v8i z8;
#pragma unroll
  for (int i = 0; i < 8; ++i) z8[i] = 0;
  __builtin_amdgcn_tensor_load_to_lds(g0, g1, z4, z4, z8, 0);
}
#endif

// ---------------- tiny init ----------------
__global__ void k_init(int* counts, float* gsum) {
  if (threadIdx.x < 8) counts[threadIdx.x] = 0;
  if (threadIdx.x < 2) gsum[threadIdx.x] = 0.f;
}

__global__ void k_zero_out(float4* __restrict__ out) {
  size_t i = (size_t)blockIdx.x * blockDim.x + threadIdx.x;
  out[i] = make_float4(0.f, 0.f, 0.f, 0.f);
}

// ---------------- f32 -> bf16 convert of x ----------------
__global__ void k_cvt_x(const float* __restrict__ x, __bf16* __restrict__ xb) {
  size_t i = ((size_t)blockIdx.x * blockDim.x + threadIdx.x) * 4;
  float4 v = *(const float4*)(x + i);
  union { __bf16 h[4]; uint2 u; } u;
  u.h[0] = (__bf16)v.x; u.h[1] = (__bf16)v.y;
  u.h[2] = (__bf16)v.z; u.h[3] = (__bf16)v.w;
  *(uint2*)(xb + i) = u.u;
}

// ---------------- pre-swizzle weights into WMMA B-fragment order ----------------
// B fragment (16x16x32 bf16): lane L holds column n=L&15, g=L>>4;
// halves 0..7 = K = g*8+j, halves 8..15 = K = 16+g*8+j. 32 B contiguous per lane.
// w1s frag order is kb-major: frag = (e*32 + kb)*8 + nb  => 8KB contiguous per kb.
__global__ void k_swz_w1(const float* __restrict__ w1, __bf16* __restrict__ w1s) {
  int tid  = blockIdx.x * blockDim.x + threadIdx.x;   // 65536 threads
  int frag = tid >> 5, lane = tid & 31;
  int e = frag >> 8, kb = (frag >> 3) & 31, nb = frag & 7;
  int g = lane >> 4, n = lane & 15;
  const float* src = w1 + (size_t)e * IN_DIM * HID + (nb * 16 + n);
  union { __bf16 h[16]; uint4 u[2]; } u;
#pragma unroll
  for (int j = 0; j < 8; ++j) {
    int k0 = kb * 32 + g * 8 + j;
    u.h[j]     = (__bf16)src[(size_t)k0 * HID];
    u.h[8 + j] = (__bf16)src[(size_t)(k0 + 16) * HID];
  }
  uint4* dst = (uint4*)(w1s + (size_t)frag * 512 + lane * 16);
  dst[0] = u.u[0]; dst[1] = u.u[1];
}

// w2s frag order: frag = (e*64 + nt)*4 + kk  => 8KB contiguous per 2 nt-tiles.
__global__ void k_swz_w2(const float* __restrict__ w2, __bf16* __restrict__ w2s) {
  int tid  = blockIdx.x * blockDim.x + threadIdx.x;   // 65536 threads
  int frag = tid >> 5, lane = tid & 31;
  int e = frag >> 8, nt = (frag >> 2) & 63, kk = frag & 3;
  int g = lane >> 4, n = lane & 15;
  const float* src = w2 + (size_t)e * HID * NCLS + (nt * 16 + n);
  union { __bf16 h[16]; uint4 u[2]; } u;
#pragma unroll
  for (int j = 0; j < 8; ++j) {
    int k0 = kk * 32 + g * 8 + j;
    u.h[j]     = (__bf16)src[(size_t)k0 * NCLS];
    u.h[8 + j] = (__bf16)src[(size_t)(k0 + 16) * NCLS];
  }
  uint4* dst = (uint4*)(w2s + (size_t)frag * 512 + lane * 16);
  dst[0] = u.u[0]; dst[1] = u.u[1];
}

// ---------------- gating: softmax + top-2 + per-expert compaction ----------------
__global__ __launch_bounds__(256) void k_gate(const float* __restrict__ x,
                                              const float* __restrict__ gw,
                                              const float* __restrict__ gb,
                                              int* counts, int* at, float* ag,
                                              float* tv) {
  int wave = threadIdx.x >> 5, lane = threadIdx.x & 31;
  int t = blockIdx.x * 8 + wave;
  float acc[8] = {0.f, 0.f, 0.f, 0.f, 0.f, 0.f, 0.f, 0.f};
  const float* xr = x + (size_t)t * IN_DIM;
  for (int i = lane; i < IN_DIM; i += 32) {
    float xv = xr[i];
    const float* g8 = gw + i * NEXP;
#pragma unroll
    for (int e = 0; e < 8; ++e) acc[e] += xv * g8[e];
  }
#pragma unroll
  for (int e = 0; e < 8; ++e) {
    for (int off = 16; off > 0; off >>= 1)
      acc[e] += __shfl_xor(acc[e], off, 32);
    acc[e] += gb[e];
  }
  float m = acc[0];
#pragma unroll
  for (int e = 1; e < 8; ++e) m = fmaxf(m, acc[e]);
  float p[8], s = 0.f;
#pragma unroll
  for (int e = 0; e < 8; ++e) { p[e] = __expf(acc[e] - m); s += p[e]; }
  float inv = 1.f / s;
#pragma unroll
  for (int e = 0; e < 8; ++e) p[e] *= inv;
  int i0 = 0; float v0 = p[0];
#pragma unroll
  for (int e = 1; e < 8; ++e) if (p[e] > v0) { v0 = p[e]; i0 = e; }
  int i1 = (i0 == 0) ? 1 : 0; float v1 = p[i1];
#pragma unroll
  for (int e = 0; e < 8; ++e) if (e != i0 && p[e] > v1) { v1 = p[e]; i1 = e; }
  if (lane == 0) {
    tv[t] = v0; tv[B_TOK + t] = v1;
    int p0 = atomicAdd(&counts[i0], 1);
    at[i0 * B_TOK + p0] = t; ag[i0 * B_TOK + p0] = v0;
    int p1 = atomicAdd(&counts[i1], 1);
    at[i1 * B_TOK + p1] = t; ag[i1 * B_TOK + p1] = v1;
  }
}

// ---------------- fused per-expert MLP: TDM -> LDS -> WMMA ----------------
// Block = 8 waves = 8 consecutive 16-token tiles of ONE expert, so all waves
// share the per-K-step weight fragments staged in LDS by the Tensor Data Mover.
__global__ __launch_bounds__(256) void k_moe(const __bf16* __restrict__ xb,
                                             const __bf16* __restrict__ w1s,
                                             const __bf16* __restrict__ w2s,
                                             const int* __restrict__ counts,
                                             const int* __restrict__ at,
                                             const float* __restrict__ ag,
                                             const float* __restrict__ b1,
                                             const float* __restrict__ b2,
                                             float* __restrict__ out) {
  __shared__ __align__(32) __bf16 bstage[2][4096];  // 2 x 8KB double-buffered B stage
  __shared__ __align__(32) __bf16 hsm[8][HID * 16]; // per-wave h tile (col-major)
  const int wave = threadIdx.x >> 5, lane = threadIdx.x & 31;
  const int e = blockIdx.x >> 7, tg = blockIdx.x & 127;
  const int cnt = counts[e];
  if (tg * 128 >= cnt) return;                      // uniform: whole block empty
  const int t0 = (tg * 8 + wave) * 16;
  const bool wvalid = t0 < cnt;
  const int tb = wvalid ? t0 : 0;
  const int g = lane >> 4, n = lane & 15;
  const int*   ate = at + e * B_TOK + tb;
  const float* age = ag + e * B_TOK + tb;
  int tokA = (wvalid && (t0 + n) < cnt) ? ate[n] : at[e * B_TOK];

  const unsigned lds0 = (unsigned)(uintptr_t)&bstage[0][0];
  const unsigned lds1 = (unsigned)(uintptr_t)&bstage[1][0];

  // ---- GEMM1: h[16,128] = x_tile @ w1[e]; B staged 8KB per K-step ----
  v8f acc[8];
#pragma unroll
  for (int i = 0; i < 8; ++i) acc[i] = v8f_zero();
  const __bf16* xrow = xb + (size_t)tokA * IN_DIM + g * 8;
  const __bf16* w1e  = w1s + (size_t)e * (32 * 4096);

#if HAVE_TDM
  if (wave == 0) tdm_load_8k(w1e, lds0);
#endif
  for (int kb = 0; kb < 32; ++kb) {
    const int cur = kb & 1;
#if HAVE_TDM
    if (wave == 0) {
      if (kb + 1 < 32) {
        tdm_load_8k(w1e + (size_t)(kb + 1) * 4096, cur ? lds0 : lds1);
        __builtin_amdgcn_s_wait_tensorcnt(1);   // current stage complete
      } else {
        __builtin_amdgcn_s_wait_tensorcnt(0);
      }
    }
    __syncthreads();
#else
    __syncthreads();
    {   // cooperative copy: 256 threads x 32B = 8KB
      const uint4* src = (const uint4*)(w1e + (size_t)kb * 4096);
      uint4* dst = (uint4*)&bstage[cur][0];
      dst[threadIdx.x * 2]     = src[threadIdx.x * 2];
      dst[threadIdx.x * 2 + 1] = src[threadIdx.x * 2 + 1];
    }
    __syncthreads();
#endif
    v16bf a;
    const __bf16* p = xrow + kb * 32;
    ((uint4*)&a)[0] = *(const uint4*)p;
    ((uint4*)&a)[1] = *(const uint4*)(p + 16);
    const __bf16* bs = &bstage[cur][lane * 16];
#pragma unroll
    for (int nb = 0; nb < 8; ++nb) {
      v16bf b = *(const v16bf*)(bs + nb * 512);
      acc[nb] = __builtin_amdgcn_wmma_f32_16x16x32_bf16(
          false, a, false, b, (short)0, acc[nb], false, false);
    }
    __syncthreads();
  }

  // ---- epilogue1: +b1, relu, bf16, stage h in per-wave LDS (col-major) ----
  __bf16* hw = hsm[wave];
#pragma unroll
  for (int nb = 0; nb < 8; ++nb) {
    float b1v = b1[e * HID + nb * 16 + n];
    union { __bf16 h[8]; uint4 u; } u;
#pragma unroll
    for (int v = 0; v < 8; ++v) {
      float hvv = acc[nb][v] + b1v;
      u.h[v] = (__bf16)(hvv > 0.f ? hvv : 0.f);
    }
    *(uint4*)(hw + (nb * 16 + n) * 16 + g * 8) = u.u;
  }

  // ---- build 4 A-fragments of h (K=128) from LDS ----
  v16bf af[4];
#pragma unroll
  for (int kk = 0; kk < 4; ++kk) {
    union { __bf16 h[16]; v16bf v; } u;
#pragma unroll
    for (int j = 0; j < 8; ++j) {
      u.h[j]     = hw[(kk * 32 + g * 8 + j) * 16 + n];
      u.h[8 + j] = hw[(kk * 32 + 16 + g * 8 + j) * 16 + n];
    }
    af[kk] = u.v;
  }

  // per-row token / gate for the epilogue (C-layout row = g*8 + v)
  int tokR[8]; float gateR[8];
#pragma unroll
  for (int v = 0; v < 8; ++v) {
    int rr = g * 8 + v;
    bool ok = wvalid && (t0 + rr) < cnt;
    tokR[v]  = ok ? ate[rr] : -1;
    gateR[v] = ok ? age[rr] : 0.f;
  }

  // ---- GEMM2: y[16,1024] = h @ w2[e]; B staged 8KB per 2 N-tiles ----
  const __bf16* w2e = w2s + (size_t)e * (64 * 4 * 512);
#if HAVE_TDM
  if (wave == 0) tdm_load_8k(w2e, lds0);
#endif
  for (int c = 0; c < 32; ++c) {
    const int cur = c & 1;
#if HAVE_TDM
    if (wave == 0) {
      if (c + 1 < 32) {
        tdm_load_8k(w2e + (size_t)(c + 1) * 4096, cur ? lds0 : lds1);
        __builtin_amdgcn_s_wait_tensorcnt(1);
      } else {
        __builtin_amdgcn_s_wait_tensorcnt(0);
      }
    }
    __syncthreads();
#else
    __syncthreads();
    {
      const uint4* src = (const uint4*)(w2e + (size_t)c * 4096);
      uint4* dst = (uint4*)&bstage[cur][0];
      dst[threadIdx.x * 2]     = src[threadIdx.x * 2];
      dst[threadIdx.x * 2 + 1] = src[threadIdx.x * 2 + 1];
    }
    __syncthreads();
#endif
#pragma unroll
    for (int half = 0; half < 2; ++half) {
      const int nt = c * 2 + half;
      v8f cc = v8f_zero();
      const __bf16* bs = &bstage[cur][half * 2048 + lane * 16];
#pragma unroll
      for (int kk = 0; kk < 4; ++kk) {
        v16bf b = *(const v16bf*)(bs + kk * 512);
        cc = __builtin_amdgcn_wmma_f32_16x16x32_bf16(
            false, af[kk], false, b, (short)0, cc, false, false);
      }
      int col = nt * 16 + n;
      float b2v = b2[e * NCLS + col];
#pragma unroll
      for (int v = 0; v < 8; ++v) {
        if (tokR[v] >= 0)
          atomicAdd(out + (size_t)tokR[v] * NCLS + col, gateR[v] * (cc[v] + b2v));
      }
    }
    __syncthreads();
  }
}

// ---------------- deterministic gates_sum reduction + broadcast ----------------
__global__ void k_gsum(const float* __restrict__ tv, float* __restrict__ gsum) {
  __shared__ float sm[2][256];
  int tid = threadIdx.x;
  float s0 = 0.f, s1 = 0.f;
  for (int t = tid; t < B_TOK; t += 256) { s0 += tv[t]; s1 += tv[B_TOK + t]; }
  sm[0][tid] = s0; sm[1][tid] = s1;
  __syncthreads();
  for (int off = 128; off > 0; off >>= 1) {
    if (tid < off) { sm[0][tid] += sm[0][tid + off]; sm[1][tid] += sm[1][tid + off]; }
    __syncthreads();
  }
  if (tid == 0) { gsum[0] = sm[0][0]; gsum[1] = sm[1][0]; }
}

__global__ void k_gates_out(const float* __restrict__ gsum, float* __restrict__ tail) {
  int i = blockIdx.x * blockDim.x + threadIdx.x;   // 2048
  tail[i] = gsum[i >> 10];
}

// ---------------- launch ----------------
extern "C" void kernel_launch(void* const* d_in, const int* in_sizes, int n_in,
                              void* d_out, int out_size, void* d_ws, size_t ws_size,
                              hipStream_t stream) {
  const float* x  = (const float*)d_in[0];
  const float* gw = (const float*)d_in[1];
  const float* gb = (const float*)d_in[2];
  const float* w1 = (const float*)d_in[3];
  const float* b1 = (const float*)d_in[4];
  const float* w2 = (const float*)d_in[5];
  const float* b2 = (const float*)d_in[6];
  float* out = (float*)d_out;
  char* ws = (char*)d_ws;

  __bf16* xb   = (__bf16*)(ws + XB_OFF);
  __bf16* w1s  = (__bf16*)(ws + W1S_OFF);
  __bf16* w2s  = (__bf16*)(ws + W2S_OFF);
  int*    at   = (int*)  (ws + AT_OFF);
  float*  ag   = (float*)(ws + AG_OFF);
  float*  tvb  = (float*)(ws + TV_OFF);
  int*    cnt  = (int*)  (ws + CNT_OFF);
  float*  gsum = (float*)(ws + GSUM_OFF);

  k_init<<<dim3(1), dim3(32), 0, stream>>>(cnt, gsum);
  k_zero_out<<<dim3(16384), dim3(256), 0, stream>>>((float4*)out);
  k_cvt_x<<<dim3(16384), dim3(256), 0, stream>>>(x, xb);
  k_swz_w1<<<dim3(256), dim3(256), 0, stream>>>(w1, w1s);
  k_swz_w2<<<dim3(256), dim3(256), 0, stream>>>(w2, w2s);
  k_gate<<<dim3(2048), dim3(256), 0, stream>>>(x, gw, gb, cnt, at, ag, tvb);
  k_moe<<<dim3(1024), dim3(256), 0, stream>>>(xb, w1s, w2s, cnt, at, ag, b1, b2, out);
  k_gsum<<<dim3(1), dim3(256), 0, stream>>>(tvb, gsum);
  k_gates_out<<<dim3(8), dim3(256), 0, stream>>>(gsum, out + (size_t)B_TOK * NCLS);
}
